// MySupConLoss_30777735643988
// MI455X (gfx1250) — matrix operand order repaired and checked by reading, not compiled, not run on hardware
//
#include <hip/hip_runtime.h>
#include <hip/hip_bf16.h>

typedef _Float16 v16h __attribute__((ext_vector_type(16)));
typedef _Float16 v8h  __attribute__((ext_vector_type(8)));
typedef float    v8f  __attribute__((ext_vector_type(8)));
typedef unsigned int v4u __attribute__((ext_vector_type(4)));
typedef int          v8i __attribute__((ext_vector_type(8)));
typedef int          v4i __attribute__((ext_vector_type(4)));

#define NROWS 8192
#define DIM   512
#define INV_T (1.0f / 0.07f)
#define NT    (NROWS / 16)
#define LDS_ROW  520            // halves per LDS row: 512 + 8 pad (16B) -> bank-conflict-free
#define LDS_TILE (16 * LDS_ROW) // halves per tile buffer (16.25 KB)

#if __has_builtin(__builtin_amdgcn_tensor_load_to_lds)
#define USE_TDM 1
#else
#define USE_TDM 0
#endif

// Build a v16h from two contiguous 16B chunks at p and p+8 halves.
static __device__ __forceinline__ v16h ld16(const _Float16* p) {
  v8h lo = *(const v8h*)p;
  v8h hi = *(const v8h*)(p + 8);
  return __builtin_shufflevector(lo, hi, 0,1,2,3,4,5,6,7,8,9,10,11,12,13,14,15);
}

// A fragment: a[0..7] = p[0..7], a[8..15] = p[16..23] (ISA 16-bit A 16x32 layout).
static __device__ __forceinline__ v16h ld_a(const _Float16* p) {
  v8h lo = *(const v8h*)p;
  v8h hi = *(const v8h*)(p + 16);
  return __builtin_shufflevector(lo, hi, 0,1,2,3,4,5,6,7,8,9,10,11,12,13,14,15);
}

#if USE_TDM
// One TDM descriptor moves a 16-row x 1024B tile of fn16 into LDS, with the
// TDM pad unit inserting 16B after every 1024B row -> 1040B LDS row stride.
static __device__ __forceinline__ void tdm_load_tile(const _Float16* gsrc, unsigned lds_off) {
  unsigned long long ga = (unsigned long long)(uintptr_t)gsrc;
  // Group0: count=1 | lds_addr | global_addr[56:0] | type=2
  v4u g0 = { 1u,
             lds_off,
             (unsigned)ga,
             (unsigned)((ga >> 32) & 0x1FFFFFFu) | (2u << 30) };
  // Group1: data_size=4B(code2), pad_enable, pad_interval=256dw(code7), pad_amount=4dw(code3)
  //         tensor_dim0=256dw, tensor_dim1=8192, tile_dim0=256dw, tile_dim1=16,
  //         tensor_dim0_stride=256dw, tensor_dim1_stride=256dw
  v8i g1 = { (int)((2u << 16) | (1u << 20) | (7u << 22) | (3u << 25)),
             (int)(256u << 16),    // tensor_dim0[15:0] in [31:16]
             (int)(8192u << 16),   // tensor_dim1[15:0] in [31:16]
             (int)(256u << 16),    // tile_dim0 in [31:16]
             16,                   // tile_dim1
             256,                  // tensor_dim0_stride[31:0]
             (int)(256u << 16),    // tensor_dim1_stride[15:0] in [31:16]
             0 };
  v4i g2 = {0,0,0,0};
  v4i g3 = {0,0,0,0};
#if defined(__clang_major__) && (__clang_major__ >= 23)
  v8i g4 = {0,0,0,0,0,0,0,0};
  __builtin_amdgcn_tensor_load_to_lds(g0, g1, g2, g3, g4, 0);
#else
  __builtin_amdgcn_tensor_load_to_lds(g0, g1, g2, g3, 0);
#endif
}
#endif

// ---------------- kernel 1: row L2-normalize, fp32 -> f16 ----------------
__global__ void k_normalize(const float* __restrict__ feat, _Float16* __restrict__ fn16) {
  const int row = blockIdx.x;
  const int t = threadIdx.x;              // 256 threads, 2 elems each
  const float* fr = feat + (size_t)row * DIM;
  float f0 = fr[t * 2 + 0];
  float f1 = fr[t * 2 + 1];
  float ss = f0 * f0 + f1 * f1;
  #pragma unroll
  for (int m = 16; m >= 1; m >>= 1) ss += __shfl_xor(ss, m, 32);
  __shared__ float smem[8];
  const int wave = t >> 5, lane = t & 31;
  if (lane == 0) smem[wave] = ss;
  __syncthreads();
  float tot = 0.f;
  #pragma unroll
  for (int w = 0; w < 8; ++w) tot += smem[w];
  const float inv = 1.0f / fmaxf(sqrtf(tot), 1e-8f);   // matches max(norm, EPS)
  _Float16* out = fn16 + (size_t)row * DIM;
  out[t * 2 + 0] = (_Float16)(f0 * inv);
  out[t * 2 + 1] = (_Float16)(f1 * inv);
}

// ---------------- kernel 2: fused GEMM + online masked softmax loss ----------------
// 4 waves / block, wave owns 16 rows. TDM double-buffers 16-col B tiles into LDS.
__global__ __launch_bounds__(128, 1)
void k_supcon(const _Float16* __restrict__ fn16,
              const int* __restrict__ pos,
              const int* __restrict__ neg,
              float* __restrict__ loss) {
#if USE_TDM
  __shared__ __attribute__((aligned(64))) _Float16 bbuf[2][LDS_TILE];
#endif
  const int lane = threadIdx.x & 31;
  const int wave = threadIdx.x >> 5;
  const int rowbase = blockIdx.x * 64 + wave * 16;
  const int half = lane >> 4;       // 0 or 1
  const int ln   = lane & 15;

  // Preload resident A fragments for this wave's 16-row strip (K = 512).
  const _Float16* aptr = fn16 + (size_t)(rowbase + ln) * DIM;
  v16h a[16];
  #pragma unroll
  for (int kt = 0; kt < 16; ++kt)
    a[kt] = ld16(aptr + kt * 32 + half * 16);

  // Online-softmax state per C/D slot: slot (r, lane) = row rowbase+r+8*half, col tile+ln.
  float m[8], sp[8], sn[8], cp[8];
  int grow[8];
  #pragma unroll
  for (int r = 0; r < 8; ++r) {
    m[r] = -1e30f; sp[r] = 0.f; sn[r] = 0.f; cp[r] = 0.f;
    grow[r] = rowbase + r + half * 8;
  }

  const int koff = half * 16;
#if USE_TDM
  const unsigned lds_base = (unsigned)(uintptr_t)&bbuf[0][0]; // flat->LDS: low 32 bits
  if (wave == 0) tdm_load_tile(fn16, lds_base);               // tile 0 -> buf 0
#endif

  for (int tile = 0; tile < NT; ++tile) {
    const int col0 = tile * 16;
#if USE_TDM
    if (wave == 0) __builtin_amdgcn_s_wait_tensorcnt(0);      // buf[tile&1] ready
    __syncthreads();                                          // publish to all waves
    asm volatile("" ::: "memory");
    if (wave == 0 && (tile + 1) < NT)                         // prefetch next tile via DMA
      tdm_load_tile(fn16 + (size_t)(tile + 1) * 16 * DIM,
                    lds_base + (unsigned)(((tile + 1) & 1) * LDS_TILE * 2));
    const _Float16* bbase = &bbuf[tile & 1][ln * LDS_ROW + koff];
#else
    const _Float16* bbase = fn16 + (size_t)(col0 + ln) * DIM + koff;
#endif

    // Issue this tile's mask loads early so their latency hides under the WMMAs.
    const int gcol = col0 + ln;
    int pv[8], nv[8];
    #pragma unroll
    for (int r = 0; r < 8; ++r) {
      const size_t idx = (size_t)grow[r] * NROWS + gcol;
      pv[r] = pos[idx];
      nv[r] = neg[idx];
    }

    // Software-pipelined K loop: 4-fragment B ring, loads 4 WMMAs ahead.
    v8f acc0 = {0.f,0.f,0.f,0.f,0.f,0.f,0.f,0.f};
    v8f acc1 = {0.f,0.f,0.f,0.f,0.f,0.f,0.f,0.f};
    v16h b[4];
    b[0] = ld16(bbase + 0 * 32);
    b[1] = ld16(bbase + 1 * 32);
    b[2] = ld16(bbase + 2 * 32);
    b[3] = ld16(bbase + 3 * 32);
    #pragma unroll
    for (int kt = 0; kt < 16; kt += 2) {
      v16h c0 = b[kt & 3];
      v16h c1 = b[(kt + 1) & 3];
      if (kt + 4 < 16) {
        b[kt & 3]       = ld16(bbase + (kt + 4) * 32);
        b[(kt + 1) & 3] = ld16(bbase + (kt + 5) * 32);
      }
      acc0 = __builtin_amdgcn_wmma_f32_16x16x32_f16(false, a[kt], false, c0,
                                                    (short)0, acc0, false, false);
      acc1 = __builtin_amdgcn_wmma_f32_16x16x32_f16(false, a[kt + 1], false, c1,
                                                    (short)0, acc1, false, false);
    }

    // Prefetch mask lines 8 tiles ahead (global_prefetch_b8).
    if (tile + 8 < NT) {
      const int pc = col0 + 128 + ln;
      #pragma unroll
      for (int r = 0; r < 8; ++r) {
        const size_t pidx = (size_t)grow[r] * NROWS + pc;
        __builtin_prefetch(pos + pidx, 0, 0);
        __builtin_prefetch(neg + pidx, 0, 0);
      }
    }

    #pragma unroll
    for (int r = 0; r < 8; ++r) {
      const bool diag = (grow[r] == gcol);
      const float pm = diag ? 0.f : (float)pv[r];
      const float nm = diag ? 0.f : (float)nv[r];
      const float s = (acc0[r] + acc1[r]) * INV_T;   // sim (diag included for max)
      const float mn = fmaxf(m[r], s);
      const float scale = __expf(m[r] - mn);
      const float e = __expf(s - mn);
      sp[r] = sp[r] * scale + e * pm;
      sn[r] = sn[r] * scale + e * nm;
      cp[r] += pm;
      m[r] = mn;
    }
#if USE_TDM
    __syncthreads();   // all waves done reading buf[tile&1] before DMA reuses it
#endif
  }

  // Exact per-row combine across the 16 slots (butterfly inside each 16-lane half).
  #pragma unroll
  for (int r = 0; r < 8; ++r) {
    float M = m[r];
    #pragma unroll
    for (int x = 1; x < 16; x <<= 1) M = fmaxf(M, __shfl_xor(M, x, 32));
    const float w = __expf(m[r] - M);
    float spT = sp[r] * w, snT = sn[r] * w, cT = cp[r];
    #pragma unroll
    for (int x = 1; x < 16; x <<= 1) {
      spT += __shfl_xor(spT, x, 32);
      snT += __shfl_xor(snT, x, 32);
      cT  += __shfl_xor(cT,  x, 32);
    }
    const float L = __logf(snT);
    const float card = (cT > 0.f) ? cT : 1.f;
    const float lrow = -(spT - L * cT) / card;
    if (ln == 0) loss[grow[r]] = lrow;
  }
}

// ---------------- kernel 3: deterministic mean reduction ----------------
__global__ void k_reduce(const float* __restrict__ loss, float* __restrict__ out) {
  __shared__ float smem[256];
  const int t = threadIdx.x;
  float s = 0.f;
  for (int i = t; i < NROWS; i += 256) s += loss[i];
  smem[t] = s;
  __syncthreads();
  for (int h = 128; h > 0; h >>= 1) {
    if (t < h) smem[t] += smem[t + h];
    __syncthreads();
  }
  if (t == 0) out[0] = smem[0] / (float)NROWS;
}

extern "C" void kernel_launch(void* const* d_in, const int* in_sizes, int n_in,
                              void* d_out, int out_size, void* d_ws, size_t ws_size,
                              hipStream_t stream) {
  (void)in_sizes; (void)n_in; (void)out_size; (void)ws_size;
  const float* feat = (const float*)d_in[0];
  const int*   pos  = (const int*)d_in[1];
  const int*   neg  = (const int*)d_in[2];
  float* out = (float*)d_out;

  _Float16* fn16 = (_Float16*)d_ws;                                  // 8 MB
  float* loss = (float*)((char*)d_ws + (size_t)NROWS * DIM * 2);     // 32 KB

  k_normalize<<<NROWS, 256, 0, stream>>>(feat, fn16);
  k_supcon<<<NROWS / 64, 128, 0, stream>>>(fn16, pos, neg, loss);
  k_reduce<<<1, 256, 0, stream>>>(loss, out);
}
